// SABlock_87119116632476
// MI455X (gfx1250) — compile-verified, matrix operand order checked by reading, not run on hardware
//
#include <hip/hip_runtime.h>
#include <hip/hip_bf16.h>
#include <hip/hip_fp16.h>

typedef __attribute__((ext_vector_type(16))) _Float16     v16h;
typedef __attribute__((ext_vector_type(8)))  float        v8f;
typedef __attribute__((ext_vector_type(4)))  unsigned int u32x4;
typedef __attribute__((ext_vector_type(8)))  int          i32x8;
typedef __attribute__((ext_vector_type(4)))  int          i32x4;

#define HIDDEN   1024
#define HEADS    16
#define HEAD_DIM 64
#define BATCH    4
#define SEQ      2048
#define ATT_SCALE 0.125f   // 64^-0.5

// B panel staged in LDS: 64 columns x K halfs, padded 8 halfs per 512 halfs
// (TDM pad_interval=7 -> 256 DWORDs, pad_amount=3 -> 4 DWORDs) to spread banks.
#define LDS_ROW_HALFS(K)  ((K) + ((K) / 512) * 8)
#define LDS_PANEL_BYTES   (64 * LDS_ROW_HALFS(1024) * 2)   // 133120 B < 320 KB

// ---------------------------------------------------------------------------
// 16-lane butterfly reductions (wave32: one WMMA C-row spans 16 lanes)
// ---------------------------------------------------------------------------
__device__ __forceinline__ float rowmax16(float v) {
    v = fmaxf(v, __shfl_xor(v, 1, 16));
    v = fmaxf(v, __shfl_xor(v, 2, 16));
    v = fmaxf(v, __shfl_xor(v, 4, 16));
    v = fmaxf(v, __shfl_xor(v, 8, 16));
    return v;
}
__device__ __forceinline__ float rowsum16(float v) {
    v += __shfl_xor(v, 1, 16);
    v += __shfl_xor(v, 2, 16);
    v += __shfl_xor(v, 4, 16);
    v += __shfl_xor(v, 8, 16);
    return v;
}

// ---------------------------------------------------------------------------
// f32 -> f16 elementwise convert
// ---------------------------------------------------------------------------
__global__ void cvt_f32_to_f16(const float* __restrict__ src,
                               _Float16* __restrict__ dst, int n) {
    int i = (blockIdx.x * blockDim.x + threadIdx.x) * 4;
    if (i + 3 < n) {
        float4 f = *(const float4*)(src + i);
        dst[i + 0] = (_Float16)f.x;
        dst[i + 1] = (_Float16)f.y;
        dst[i + 2] = (_Float16)f.z;
        dst[i + 3] = (_Float16)f.w;
    } else {
        for (; i < n; ++i) dst[i] = (_Float16)src[i];
    }
}

// ---------------------------------------------------------------------------
// TDM: stage a 64-row x K-half B panel (row-major [N,K] weights) into LDS
// with bank-spreading padding. Issued by one wave; D# per ISA 08_async_tensor.
// ---------------------------------------------------------------------------
__device__ __forceinline__ void tdm_stage_panel(const _Float16* gsrc,
                                                unsigned lds_base, int K) {
    const unsigned long long ga = (unsigned long long)(uintptr_t)gsrc;
    u32x4 g0 = { 1u,                                   // count=1, user desc
                 lds_base,                             // lds_addr [63:32]
                 (unsigned)(ga & 0xffffffffu),         // global_addr lo
                 (unsigned)((ga >> 32) & 0x01ffffffu)  // global_addr hi (57b)
                   | (2u << 30) };                     // type=2 (image)
    i32x8 g1;
    g1[0] = (int)((1u << 16)        // data_size = 1 -> 2 bytes
                | (1u << 20)        // pad_enable
                | (7u << 22)        // pad_interval = 7 -> 256 DWORDs
                | (3u << 25));      // pad_amount   = 3 -> 4 DWORDs (8 halfs)
    g1[1] = (int)((unsigned)(K & 0xffff) << 16);          // tensor_dim0 lo16
    g1[2] = (int)(((unsigned)K >> 16) | (64u << 16));     // dim0 hi | dim1 lo
    g1[3] = (int)((unsigned)(K & 0xffff) << 16);          // tile_dim0 = K
    g1[4] = 64;                                           // tile_dim1 = 64
    g1[5] = K;                                            // dim0_stride lo32
    g1[6] = 0;
    g1[7] = 0;
    i32x4 gz4 = { 0, 0, 0, 0 };
    i32x8 gz8 = { 0, 0, 0, 0, 0, 0, 0, 0 };
    __builtin_amdgcn_tensor_load_to_lds(g0, g1, gz4, gz4, gz8, 0);
    __builtin_amdgcn_s_wait_tensorcnt(0);
}

// LDS half-offset for element (col c, k-offset koff) of the padded panel
__device__ __forceinline__ int lds_panel_off(int c, int koff, int K) {
    return c * LDS_ROW_HALFS(K) + koff + ((koff >> 9) << 3);  // +8 per 512
}

// ---------------------------------------------------------------------------
// WMMA GEMM: C[M,N] = A[M,K] @ B^T  (B stored [N,K] row-major).
// block = 8 waves, tile 256(M) x 64(N); wave = 32(M) x 64(N) -> 8 WMMA/k-step.
// B panel lives in LDS (TDM-staged); A streams from global.
// MODE 0: scatter f16 -> Q[bh][s][d], K[bh][s][d], Vt[bh][d][s]
// MODE 1: f32 + bias -> Out
// ---------------------------------------------------------------------------
template <int MODE>
__global__ __launch_bounds__(256)
void gemm_wmma(const _Float16* __restrict__ A, const _Float16* __restrict__ Bw,
               int M, int N, int K,
               _Float16* __restrict__ Qo, _Float16* __restrict__ Ko,
               _Float16* __restrict__ Vt,
               float* __restrict__ Out, const float* __restrict__ bias) {
    extern __shared__ _Float16 ldsB[];

    const int wave  = threadIdx.x >> 5;
    const int lane  = threadIdx.x & 31;
    const int laneN = lane & 15;
    const int hi    = lane >> 4;
    const int mBase = blockIdx.x * 256 + wave * 32;
    const int nBase = blockIdx.y * 64;

    // one wave DMAs the 64 x K B panel into LDS, then the block syncs
    if (wave == 0)
        tdm_stage_panel(Bw + (size_t)nBase * K, (unsigned)(uintptr_t)ldsB, K);
    __syncthreads();

    const size_t aRow0 = (size_t)(mBase + laneN) * K + hi * 16;
    const size_t aRow1 = (size_t)(mBase + 16 + laneN) * K + hi * 16;

    v8f acc[2][4] = {};

    for (int k0 = 0; k0 < K; k0 += 32) {
        v16h af0 = *(const v16h*)(A + aRow0 + k0);
        v16h af1 = *(const v16h*)(A + aRow1 + k0);
        __builtin_prefetch(A + aRow0 + k0 + 64, 0, 3);   // global_prefetch_b8
        __builtin_prefetch(A + aRow1 + k0 + 64, 0, 3);
#pragma unroll
        for (int a = 0; a < 4; ++a) {
            const v16h bf = *(const v16h*)(ldsB +
                lds_panel_off(a * 16 + laneN, k0 + hi * 16, K));
            acc[0][a] = __builtin_amdgcn_wmma_f32_16x16x32_f16(
                false, af0, false, bf, (short)0, acc[0][a], false, false);
            acc[1][a] = __builtin_amdgcn_wmma_f32_16x16x32_f16(
                false, af1, false, bf, (short)0, acc[1][a], false, false);
        }
    }

#pragma unroll
    for (int sub = 0; sub < 2; ++sub) {
#pragma unroll
        for (int a = 0; a < 4; ++a) {
#pragma unroll
            for (int j = 0; j < 8; ++j) {
                const int m   = mBase + sub * 16 + j + hi * 8;
                const int n   = nBase + a * 16 + laneN;
                const float v = acc[sub][a][j];
                if (MODE == 0) {
                    const int i = n >> 10, r = n & 1023;
                    const int h = r >> 6,  d = r & 63;
                    const int b = m >> 11, s = m & 2047;
                    const size_t bh = (size_t)b * HEADS + h;
                    if (i == 0)
                        Qo[(bh * SEQ + s) * HEAD_DIM + d] = (_Float16)v;
                    else if (i == 1)
                        Ko[(bh * SEQ + s) * HEAD_DIM + d] = (_Float16)v;
                    else
                        Vt[(bh * HEAD_DIM + d) * SEQ + s] = (_Float16)v;
                } else {
                    Out[(size_t)m * N + n] = v + bias[n];
                }
            }
        }
    }
}

// ---------------------------------------------------------------------------
// Flash attention: grid (B*HEADS, SEQ/128), block 256 (8 waves x 16 q-rows)
// key chunks of 32; online softmax; P re-layout C->A through per-wave LDS
// (same-wave LDS ops are in-order on CDNA5 -> no barrier needed)
// ---------------------------------------------------------------------------
__global__ __launch_bounds__(256)
void flash_attn(const _Float16* __restrict__ Qh, const _Float16* __restrict__ Kh,
                const _Float16* __restrict__ Vt, _Float16* __restrict__ O) {
    __shared__ _Float16 ldsP[8][16 * 32];    // 8 KB: per-wave 16x32 P tile

    const int wave  = threadIdx.x >> 5;
    const int lane  = threadIdx.x & 31;
    const int laneN = lane & 15;
    const int hi    = lane >> 4;
    const int bh    = blockIdx.x;
    const int b     = bh >> 4;
    const int h     = bh & 15;
    const int q0    = blockIdx.y * 128 + wave * 16;

    const _Float16* Qb = Qh + (size_t)bh * SEQ * HEAD_DIM;
    const _Float16* Kb = Kh + (size_t)bh * SEQ * HEAD_DIM;
    const _Float16* Vb = Vt + (size_t)bh * HEAD_DIM * SEQ;

    const v16h qf0 = *(const v16h*)(Qb + (size_t)(q0 + laneN) * HEAD_DIM + hi * 16);
    const v16h qf1 = *(const v16h*)(Qb + (size_t)(q0 + laneN) * HEAD_DIM + 32 + hi * 16);

    float mrow[8], lrow[8];
#pragma unroll
    for (int j = 0; j < 8; ++j) { mrow[j] = -1e30f; lrow[j] = 0.0f; }
    v8f o[4] = {};

    _Float16* P = &ldsP[wave][0];

    for (int key0 = 0; key0 < SEQ; key0 += 32) {
        // ---- S = Q @ K^T, 16x32 score tile
        v8f s0 = {}, s1 = {};
        {
            v16h k00 = *(const v16h*)(Kb + (size_t)(key0 + laneN)      * HEAD_DIM + hi * 16);
            v16h k01 = *(const v16h*)(Kb + (size_t)(key0 + laneN)      * HEAD_DIM + 32 + hi * 16);
            v16h k10 = *(const v16h*)(Kb + (size_t)(key0 + 16 + laneN) * HEAD_DIM + hi * 16);
            v16h k11 = *(const v16h*)(Kb + (size_t)(key0 + 16 + laneN) * HEAD_DIM + 32 + hi * 16);
            s0 = __builtin_amdgcn_wmma_f32_16x16x32_f16(false, qf0, false, k00, (short)0, s0, false, false);
            s0 = __builtin_amdgcn_wmma_f32_16x16x32_f16(false, qf1, false, k01, (short)0, s0, false, false);
            s1 = __builtin_amdgcn_wmma_f32_16x16x32_f16(false, qf0, false, k10, (short)0, s1, false, false);
            s1 = __builtin_amdgcn_wmma_f32_16x16x32_f16(false, qf1, false, k11, (short)0, s1, false, false);
        }

        // ---- online softmax per row
        float p0[8], p1[8], alpha[8];
#pragma unroll
        for (int j = 0; j < 8; ++j) {
            const float a0 = s0[j] * ATT_SCALE;
            const float a1 = s1[j] * ATT_SCALE;
            const float mx = rowmax16(fmaxf(a0, a1));
            const float mn = fmaxf(mrow[j], mx);
            alpha[j] = __expf(mrow[j] - mn);
            p0[j]    = __expf(a0 - mn);
            p1[j]    = __expf(a1 - mn);
            lrow[j]  = lrow[j] * alpha[j] + rowsum16(p0[j] + p1[j]);
            mrow[j]  = mn;
        }
#pragma unroll
        for (int a = 0; a < 4; ++a)
#pragma unroll
            for (int j = 0; j < 8; ++j) o[a][j] *= alpha[j];

        // ---- P: C-layout -> LDS -> A-fragment (same-wave LDS is in-order)
#pragma unroll
        for (int j = 0; j < 8; ++j) {
            const int r = j + hi * 8;
            P[r * 32 + laneN]      = (_Float16)p0[j];
            P[r * 32 + 16 + laneN] = (_Float16)p1[j];
        }
        const v16h pf = *(const v16h*)(P + laneN * 32 + hi * 16);

        // ---- O += P @ V  (V transposed: contiguous along keys)
#pragma unroll
        for (int a = 0; a < 4; ++a) {
            v16h vf = *(const v16h*)(Vb + (size_t)(a * 16 + laneN) * SEQ + key0 + hi * 16);
            o[a] = __builtin_amdgcn_wmma_f32_16x16x32_f16(false, pf, false, vf,
                                                          (short)0, o[a], false, false);
        }
    }

#pragma unroll
    for (int j = 0; j < 8; ++j) lrow[j] = 1.0f / lrow[j];
#pragma unroll
    for (int a = 0; a < 4; ++a)
#pragma unroll
        for (int j = 0; j < 8; ++j) {
            const int r = q0 + j + hi * 8;
            const int c = h * HEAD_DIM + a * 16 + laneN;
            O[((size_t)b * SEQ + r) * HIDDEN + c] = (_Float16)(o[a][j] * lrow[j]);
        }
}

// ---------------------------------------------------------------------------
// host launcher
// ---------------------------------------------------------------------------
extern "C" void kernel_launch(void* const* d_in, const int* in_sizes, int n_in,
                              void* d_out, int out_size, void* d_ws, size_t ws_size,
                              hipStream_t stream) {
    (void)in_sizes; (void)n_in; (void)out_size; (void)ws_size;

    const float* x     = (const float*)d_in[0];   // [4,2048,1024]
    const float* w_qkv = (const float*)d_in[1];   // [3072,1024]
    const float* w_out = (const float*)d_in[2];   // [1024,1024]
    const float* b_out = (const float*)d_in[3];   // [1024]
    float*       out   = (float*)d_out;           // [4,2048,1024]

    const int M = BATCH * SEQ;                    // 8192

    char* ws = (char*)d_ws;
    _Float16* Xh    = (_Float16*)(ws);                      size_t off = (size_t)M * HIDDEN * 2;
    _Float16* Wqkvh = (_Float16*)(ws + off);                off += (size_t)3 * HIDDEN * HIDDEN * 2;
    _Float16* Wouth = (_Float16*)(ws + off);                off += (size_t)HIDDEN * HIDDEN * 2;
    _Float16* Qh    = (_Float16*)(ws + off);                off += (size_t)M * HIDDEN * 2;
    _Float16* Kh    = (_Float16*)(ws + off);                off += (size_t)M * HIDDEN * 2;
    _Float16* Vth   = (_Float16*)(ws + off);                off += (size_t)M * HIDDEN * 2;
    _Float16* AttOh = (_Float16*)(ws + off);

    // 1) f32 -> f16 conversions
    {
        int n;
        n = M * HIDDEN;
        cvt_f32_to_f16<<<(n / 4 + 255) / 256, 256, 0, stream>>>(x, Xh, n);
        n = 3 * HIDDEN * HIDDEN;
        cvt_f32_to_f16<<<(n / 4 + 255) / 256, 256, 0, stream>>>(w_qkv, Wqkvh, n);
        n = HIDDEN * HIDDEN;
        cvt_f32_to_f16<<<(n / 4 + 255) / 256, 256, 0, stream>>>(w_out, Wouth, n);
    }

    // 2) QKV projection (TDM-staged weights, 8 WMMA per wave per k-step)
    gemm_wmma<0><<<dim3(M / 256, (3 * HIDDEN) / 64), 256, LDS_PANEL_BYTES, stream>>>(
        Xh, Wqkvh, M, 3 * HIDDEN, HIDDEN, Qh, Kh, Vth, nullptr, nullptr);

    // 3) flash attention -> AttOh [8192,1024] f16
    flash_attn<<<dim3(BATCH * HEADS, SEQ / 128), 256, 0, stream>>>(Qh, Kh, Vth, AttOh);

    // 4) output projection + bias -> f32 d_out
    gemm_wmma<1><<<dim3(M / 256, HIDDEN / 64), 256, LDS_PANEL_BYTES, stream>>>(
        AttOh, Wouth, M, HIDDEN, HIDDEN, nullptr, nullptr, nullptr, out, b_out);
}
